// CausalSelfAttention_78005196030000
// MI455X (gfx1250) — matrix operand order repaired and checked by reading, not compiled, not run on hardware
//
#include <hip/hip_runtime.h>
#include <hip/hip_bf16.h>
#include <math.h>

typedef __attribute__((ext_vector_type(16))) __bf16 v16bf;
typedef __attribute__((ext_vector_type(8)))  float  v8f;

// ---------------------------------------------------------------------------
// GEMM: C[M,N] = A[M,K] @ Bw[N,K]^T   (bf16 WMMA 16x16x32, f32 accumulate)
//
// Block = 256 threads = 8 waves, block tile 64x128, wave tile 16x64.
// Double-buffered LDS pipeline: cooperative coalesced global loads of the
// NEXT K-step (f32 -> regs) overlap WMMA compute on the CURRENT K-step
// (bf16 LDS -> frags). f32->bf16 conversion at the LDS-store point.
//
// WMMA register layouts per CDNA5 ISA 7.12.2:
//   A 16x32 bf16 : lane L -> row (L&15); elems 0..7 -> K=(L>>4)*8+i,
//                  elems 8..15 -> K=16+(L>>4)*8+(i-8)
//   B 32x16 bf16 : lane L -> col (L&15); elem i -> same K as A elem i
//   C/D 16x16 f32: lane L -> col (L&15); VGPR r -> row r+(L>>4)*8
// ---------------------------------------------------------------------------
__global__ __launch_bounds__(256) void gemm_xWt_wmma(
    const float* __restrict__ A, const float* __restrict__ Bw,
    float* __restrict__ C, int M, int N, int K, int tilesN128)
{
  const int tid  = threadIdx.x;
  const int wid  = tid >> 5;
  const int lane = tid & 31;
  const int lq   = lane & 15;
  const int hs   = lane >> 4;
  const int mw   = wid & 3;          // 4 wave-rows of 16
  const int nw   = wid >> 2;         // 2 wave-cols of 64

  const int blockM = blockIdx.x / tilesN128;
  const int blockN = blockIdx.x % tilesN128;
  const int row0A  = blockM * 64;
  const int col0B  = blockN * 128;

  __shared__ __bf16 sA[2][64][32];    // 4 KB per buffer
  __shared__ __bf16 sB[2][128][32];   // 8 KB per buffer

  // cooperative-load coordinates (coalesced, contiguous per thread)
  const int rowA = tid >> 2;          // 0..63
  const int ka   = (tid & 3) * 8;     // 8 contiguous f32
  const int rowB = tid >> 1;          // 0..127
  const int kb   = (tid & 1) * 16;    // 16 contiguous f32

  const float* aPtr = A  + (size_t)(row0A + rowA) * K + ka;
  const float* bPtr = Bw + (size_t)(col0B + rowB) * K + kb;

  float a_pf[8], b_pf[16];

  // ---- prologue: fetch K-step 0 and stage it ----
  #pragma unroll
  for (int i = 0; i < 8;  ++i) a_pf[i] = aPtr[i];
  #pragma unroll
  for (int i = 0; i < 16; ++i) b_pf[i] = bPtr[i];
  #pragma unroll
  for (int i = 0; i < 8;  ++i) sA[0][rowA][ka + i] = (__bf16)a_pf[i];
  #pragma unroll
  for (int i = 0; i < 16; ++i) sB[0][rowB][kb + i] = (__bf16)b_pf[i];
  __syncthreads();

  v8f zacc = {};
  v8f acc[4];
  #pragma unroll
  for (int t = 0; t < 4; ++t) acc[t] = zacc;

  int buf = 0;
  for (int kk = 0; kk < K; kk += 32) {
    const bool more = (kk + 32) < K;
    // ---- issue next tile's global loads (independent regs, no waits yet) ----
    if (more) {
      #pragma unroll
      for (int i = 0; i < 8;  ++i) a_pf[i] = aPtr[kk + 32 + i];
      #pragma unroll
      for (int i = 0; i < 16; ++i) b_pf[i] = bPtr[kk + 32 + i];
    }
    // ---- compute current K-step from LDS ----
    v16bf a;
    #pragma unroll
    for (int i = 0; i < 8; ++i) a[i]     = sA[buf][mw*16 + lq][     hs*8 + i];
    #pragma unroll
    for (int i = 0; i < 8; ++i) a[8 + i] = sA[buf][mw*16 + lq][16 + hs*8 + i];
    #pragma unroll
    for (int t = 0; t < 4; ++t) {
      v16bf b;
      #pragma unroll
      for (int i = 0; i < 16; ++i) b[i] = sB[buf][nw*64 + t*16 + lq][(i < 8) ? (hs*8 + i) : (16 + hs*8 + (i - 8))];
      acc[t] = __builtin_amdgcn_wmma_f32_16x16x32_bf16(
                 false, a, false, b, (short)0, acc[t], false, false);
    }
    // ---- stage next tile into the other buffer ----
    if (more) {
      #pragma unroll
      for (int i = 0; i < 8;  ++i) sA[buf ^ 1][rowA][ka + i] = (__bf16)a_pf[i];
      #pragma unroll
      for (int i = 0; i < 16; ++i) sB[buf ^ 1][rowB][kb + i] = (__bf16)b_pf[i];
      __syncthreads();
      buf ^= 1;
    }
  }

  // ---- epilogue ----
  const int row0 = row0A + mw * 16;
  const int col0 = col0B + nw * 64;
  #pragma unroll
  for (int t = 0; t < 4; ++t) {
    #pragma unroll
    for (int r = 0; r < 8; ++r) {
      C[(size_t)(row0 + hs*8 + r) * N + col0 + t*16 + lq] = acc[t][r];
    }
  }
}

// ---------------------------------------------------------------------------
// RoPE applied in place to q and k inside the qkv buffer (B,T,3C).
// start_pos is read from device memory (graph-capture safe).
// ---------------------------------------------------------------------------
__global__ __launch_bounds__(256) void rope_kernel(
    float* __restrict__ qkv, const int* __restrict__ start_pos_p,
    int B, int T, int H, int Dh)
{
  const int half = Dh >> 1;
  int idx = blockIdx.x * blockDim.x + threadIdx.x;
  const int total = B * T * H * half;
  if (idx >= total) return;

  const int d  = idx % half;  idx /= half;
  const int hh = idx % H;     idx /= H;
  const int t  = idx % T;
  const int b  = idx / T;
  const int Cc = H * Dh, C3 = 3 * Cc;

  const float inv = __powf(10000.0f, -(float)d / (float)half);
  const float ang = (float)(start_pos_p[0] + t) * inv;
  float s, c;
  __sincosf(ang, &s, &c);

  const size_t base = ((size_t)b * T + t) * C3 + hh * Dh;
  float q1 = qkv[base + d], q2 = qkv[base + d + half];
  qkv[base + d]        = q1 * c - q2 * s;
  qkv[base + d + half] = q1 * s + q2 * c;

  const size_t kbse = base + Cc;
  float k1 = qkv[kbse + d], k2 = qkv[kbse + d + half];
  qkv[kbse + d]        = k1 * c - k2 * s;
  qkv[kbse + d + half] = k1 * s + k2 * c;
}

// ---------------------------------------------------------------------------
// Flash attention (causal, online softmax). Block = 128 threads = 4 waves,
// one (b, h, 64-query-row) tile per block; each wave owns 16 query rows.
// K/V tiles staged cooperatively through LDS (V stored transposed so P@V
// B-frags are contiguous DS reads). Per 32-key tile per active wave:
// 8 WMMA (S = Q@K^T) + 8 WMMA (O += P@V). P transposed C-layout->A-layout
// through per-wave LDS guarded by s_wait_dscnt (wave-synchronous).
// ---------------------------------------------------------------------------
__global__ __launch_bounds__(128) void attn_flash_wmma(
    const float* __restrict__ qkv, float* __restrict__ out,
    int B, int T, int H, float scale)
{
  const int Dh = 128, Cc = H * Dh, C3 = 3 * Cc;
  const int tid  = threadIdx.x;
  const int wid  = tid >> 5;
  const int lane = tid & 31;
  const int lq   = lane & 15;
  const int hs   = lane >> 4;

  const int tq = blockIdx.x % (T / 64);
  int tmp      = blockIdx.x / (T / 64);
  const int h  = tmp % H;
  const int b  = tmp / H;
  const int q0 = tq * 64 + wid * 16;

  __shared__ __bf16 sK[32][128];      // [key][d]   8 KB
  __shared__ __bf16 sVt[128][32];     // [d][key]   8 KB (transposed)
  __shared__ float  ldsP[4][16][32];  // per-wave P scratch, 8 KB

  // ---- preload this wave's 16x128 Q tile as 4 bf16 A-frags ----
  const float* qrow = qkv + ((size_t)b * T + (q0 + lq)) * C3 + h * Dh;
  v16bf aq[4];
  #pragma unroll
  for (int kp = 0; kp < 4; ++kp) {
    #pragma unroll
    for (int i = 0; i < 8; ++i) aq[kp][i]     = (__bf16)qrow[kp*32 +      hs*8 + i];
    #pragma unroll
    for (int i = 0; i < 8; ++i) aq[kp][8 + i] = (__bf16)qrow[kp*32 + 16 + hs*8 + i];
  }

  float mrun[8], lrun[8];
  #pragma unroll
  for (int r = 0; r < 8; ++r) { mrun[r] = -__builtin_inff(); lrun[r] = 0.0f; }
  v8f zacc = {};
  v8f oacc[8];
  #pragma unroll
  for (int ct = 0; ct < 8; ++ct) oacc[ct] = zacc;

  const float* kbase = qkv + (size_t)b * T * C3 +     Cc + h * Dh;
  const float* vbase = qkv + (size_t)b * T * C3 + 2 * Cc + h * Dh;

  const int kv_end_w = q0 + 16;                  // this wave's causal bound
  const int ntiles   = (tq * 64 + 64) / 32;      // block-uniform trip count

  for (int tile = 0; tile < ntiles; ++tile) {
    const int s0 = tile * 32;

    __syncthreads();   // previous compute done before overwriting K/V tiles
    // ---- cooperative K/V tile load: 16 independent b128 loads in flight ----
    float4 kreg[8], vreg[8];
    #pragma unroll
    for (int i = 0; i < 8; ++i) {
      const int f4  = i * 128 + tid;             // 1024 float4 per tile
      const int key = f4 >> 5;                   // 32 float4 per row
      const int seg = f4 & 31;
      kreg[i] = *(const float4*)(kbase + (size_t)(s0 + key) * C3 + seg * 4);
      vreg[i] = *(const float4*)(vbase + (size_t)(s0 + key) * C3 + seg * 4);
    }
    #pragma unroll
    for (int i = 0; i < 8; ++i) {
      const int f4  = i * 128 + tid;
      const int key = f4 >> 5;
      const int seg = f4 & 31;
      sK[key][seg*4 + 0] = (__bf16)kreg[i].x;
      sK[key][seg*4 + 1] = (__bf16)kreg[i].y;
      sK[key][seg*4 + 2] = (__bf16)kreg[i].z;
      sK[key][seg*4 + 3] = (__bf16)kreg[i].w;
      sVt[seg*4 + 0][key] = (__bf16)vreg[i].x;
      sVt[seg*4 + 1][key] = (__bf16)vreg[i].y;
      sVt[seg*4 + 2][key] = (__bf16)vreg[i].z;
      sVt[seg*4 + 3][key] = (__bf16)vreg[i].w;
    }
    __syncthreads();

    if (s0 < kv_end_w) {     // wave-uniform: masked waves skip compute
      // ---- S = Q @ K^T : two 16x16 tiles over 32 keys ----
      v8f sacc[2];
      sacc[0] = zacc; sacc[1] = zacc;
      #pragma unroll
      for (int nt = 0; nt < 2; ++nt) {
        #pragma unroll
        for (int kp = 0; kp < 4; ++kp) {
          v16bf bk;
          #pragma unroll
          for (int i = 0; i < 16; ++i)
            bk[i] = sK[nt*16 + lq][kp*32 + ((i < 8) ? (hs*8 + i) : (16 + hs*8 + (i - 8)))];
          sacc[nt] = __builtin_amdgcn_wmma_f32_16x16x32_bf16(
                       false, aq[kp], false, bk, (short)0, sacc[nt], false, false);
        }
      }

      // ---- causal mask + online softmax (rows live in a 16-lane half) ----
      #pragma unroll
      for (int r = 0; r < 8; ++r) {
        const int row = q0 + hs*8 + r;
        float s0v = sacc[0][r] * scale;
        float s1v = sacc[1][r] * scale;
        if (s0 +      lq > row) s0v = -__builtin_inff();
        if (s0 + 16 + lq > row) s1v = -__builtin_inff();

        float mx = fmaxf(s0v, s1v);
        mx = fmaxf(mx, __shfl_xor(mx, 1, 32));
        mx = fmaxf(mx, __shfl_xor(mx, 2, 32));
        mx = fmaxf(mx, __shfl_xor(mx, 4, 32));
        mx = fmaxf(mx, __shfl_xor(mx, 8, 32));

        const float mnew  = fmaxf(mrun[r], mx);
        const float alpha = __expf(mrun[r] - mnew);
        mrun[r] = mnew;

        const float p0 = __expf(s0v - mnew);
        const float p1 = __expf(s1v - mnew);
        float ps = p0 + p1;
        ps += __shfl_xor(ps, 1, 32);
        ps += __shfl_xor(ps, 2, 32);
        ps += __shfl_xor(ps, 4, 32);
        ps += __shfl_xor(ps, 8, 32);
        lrun[r] = lrun[r] * alpha + ps;

        #pragma unroll
        for (int ct = 0; ct < 8; ++ct) oacc[ct][r] *= alpha;

        ldsP[wid][hs*8 + r][lq]      = p0;   // C-layout -> LDS
        ldsP[wid][hs*8 + r][16 + lq] = p1;
      }

      // wave-synchronous LDS transpose: all lanes' DS stores complete
      asm volatile("s_wait_dscnt 0x0" ::: "memory");

      v16bf pa;                              // P as 16x32 bf16 A-frag
      #pragma unroll
      for (int i = 0; i < 8; ++i) pa[i]     = (__bf16)ldsP[wid][lq][     hs*8 + i];
      #pragma unroll
      for (int i = 0; i < 8; ++i) pa[8 + i] = (__bf16)ldsP[wid][lq][16 + hs*8 + i];

      // ---- O += P @ V : 8 column tiles of 16 over D=128 ----
      #pragma unroll
      for (int ct = 0; ct < 8; ++ct) {
        v16bf bv;
        #pragma unroll
        for (int i = 0; i < 16; ++i)
          bv[i] = sVt[ct*16 + lq][(i < 8) ? (hs*8 + i) : (16 + hs*8 + (i - 8))];
        oacc[ct] = __builtin_amdgcn_wmma_f32_16x16x32_bf16(
                     false, pa, false, bv, (short)0, oacc[ct], false, false);
      }
    }
  }

  // ---- epilogue: normalize and store (B,T,C) with feature = h*Dh + d ----
  #pragma unroll
  for (int ct = 0; ct < 8; ++ct) {
    #pragma unroll
    for (int r = 0; r < 8; ++r) {
      const int row = q0 + hs*8 + r;
      out[((size_t)b * T + row) * Cc + h * Dh + ct*16 + lq] = oacc[ct][r] / lrun[r];
    }
  }
}

// ---------------------------------------------------------------------------
extern "C" void kernel_launch(void* const* d_in, const int* in_sizes, int n_in,
                              void* d_out, int out_size, void* d_ws, size_t ws_size,
                              hipStream_t stream)
{
  const float* x         = (const float*)d_in[0];
  const float* Wqkv      = (const float*)d_in[1];
  const float* Wproj     = (const float*)d_in[2];
  const int*   start_pos = (const int*)d_in[3];

  const int B = 2, T = 2048, C = 2048, H = 16;
  const int Dh = C / H;          // 128
  const int M  = B * T;          // 4096
  const int N1 = 3 * C;          // 6144
  const int K  = C;

  float* qkv = (float*)d_ws;                                           // M*N1 f32 (96 MB)
  float* att = (float*)((char*)d_ws + (size_t)M * N1 * sizeof(float)); // M*C  f32 (32 MB)

  // 1) qkv = x @ Wqkv^T
  {
    const int tilesN128 = N1 / 128;                       // 48
    gemm_xWt_wmma<<<(M / 64) * tilesN128, 256, 0, stream>>>(
        x, Wqkv, qkv, M, N1, K, tilesN128);
  }
  // 2) RoPE in place on q,k
  {
    const int total = B * T * H * (Dh / 2);
    rope_kernel<<<(total + 255) / 256, 256, 0, stream>>>(qkv, start_pos, B, T, H, Dh);
  }
  // 3) causal flash attention
  {
    const float scale = 1.0f / sqrtf((float)Dh);
    attn_flash_wmma<<<B * H * (T / 64), 128, 0, stream>>>(qkv, att, B, T, H, scale);
  }
  // 4) out = att @ Wproj^T
  {
    const int tilesN128 = C / 128;                        // 16
    gemm_xWt_wmma<<<(M / 64) * tilesN128, 256, 0, stream>>>(
        att, Wproj, (float*)d_out, M, C, K, tilesN128);
  }
}
